// TrajectoryLSTM_70437463654607
// MI455X (gfx1250) — matrix operand order repaired and checked by reading, not compiled, not run on hardware
//
#include <hip/hip_runtime.h>
#include <math.h>

typedef __attribute__((ext_vector_type(16))) _Float16 v16h;
typedef __attribute__((ext_vector_type(8)))  _Float16 h8;
typedef __attribute__((ext_vector_type(8)))  float    v8f;
typedef __attribute__((ext_vector_type(4)))  float    f4;
typedef __attribute__((ext_vector_type(4)))  unsigned int u32x4;
typedef __attribute__((ext_vector_type(8)))  int      i32x8;
typedef __attribute__((ext_vector_type(4)))  int      i32x4;

#define TT 8
#define PRED 12
#define BN_EPS 1e-5f
#define WAVES 4
#define TPB (WAVES * 32)

// packed-weight region copied to LDS: wfih(32KB) wfhh(32KB) wfse2(4KB) wfhp1(2KB) blstm(1KB)
#define WREG_BYTES 72704
#define WREG_HALVES (WREG_BYTES / 2)

#if __has_builtin(__builtin_amdgcn_tensor_load_to_lds)
#define HAVE_TDM 1
#endif

// ---------------------------------------------------------------- helpers

__device__ __forceinline__ float sigm(float x) { return 1.f / (1.f + __expf(-x)); }

__device__ __forceinline__ v8f wmma32(v16h a, v16h b, v8f c) {
  return __builtin_amdgcn_wmma_f32_16x16x32_f16(false, a, false, b, (short)0, c, false, false);
}

// wave-private LDS stage fence: all ds ops complete + no compiler reordering
__device__ __forceinline__ void wave_lds_fence() {
  asm volatile("s_wait_dscnt 0" ::: "memory");
  __builtin_amdgcn_wave_barrier();
}

// B-fragment (32x16 KxN, f16): lane's 16 halves contiguous (works for global or LDS src)
__device__ __forceinline__ v16h load_bfrag(const _Float16* f, int lane) {
  union { h8 q[2]; v16h v; } u;
  const h8* p = (const h8*)(f + lane * 16);
  u.q[0] = p[0];
  u.q[1] = p[1];
  return u.v;
}

// A-fragment (16x32 MxK) from row-major fp32 matrix, convert to f16.
__device__ __forceinline__ v16h loadA_f32(const float* M, int stride, int m0, int k0, int lane) {
  int row = m0 + (lane & 15);
  int kb  = k0 + ((lane >> 4) << 3);
  const f4* p = (const f4*)(M + (size_t)row * stride + kb);
  f4 a0 = p[0], a1 = p[1];   // K kb..kb+7
  f4 b0 = p[4], b1 = p[5];   // K kb+16..kb+23
  v16h r;
  #pragma unroll
  for (int i = 0; i < 4; ++i) {
    r[i]      = (_Float16)a0[i];
    r[4 + i]  = (_Float16)a1[i];
    r[8 + i]  = (_Float16)b0[i];
    r[12 + i] = (_Float16)b1[i];
  }
  return r;
}

// A-fragment from row-major f16 LDS stage (16x64)
__device__ __forceinline__ v16h loadA_h16(const _Float16* S, int k0, int lane) {
  int row = lane & 15;
  int kb  = k0 + ((lane >> 4) << 3);
  const h8* p = (const h8*)(S + row * 64 + kb);
  union { h8 h[2]; v16h v; } u;
  u.h[0] = p[0];
  u.h[1] = p[2];   // +16 halves
  return u.v;
}

// store a C-layout 16x16 f32 tile as f16 into row-major 16x64 LDS stage
__device__ __forceinline__ void store_tile_h16(_Float16* S, int jt, v8f acc, int lane) {
  int col   = jt * 16 + (lane & 15);
  int rbase = (lane >> 4) << 3;
  #pragma unroll
  for (int r = 0; r < 8; ++r)
    S[(rbase + r) * 64 + col] = (_Float16)acc[r];
}

// One-shot TDM copy of the packed weight region into LDS (all threads must call).
__device__ __forceinline__ void load_weights_to_lds(const _Float16* gsrc, _Float16* wbuf) {
#ifdef HAVE_TDM
  if ((threadIdx.x >> 5) == 0) {   // one wave issues the DMA
    unsigned long long ga = (unsigned long long)(uintptr_t)gsrc;
    unsigned la = (unsigned)(uintptr_t)(void*)wbuf;   // low 32 bits = LDS offset
    unsigned E = WREG_BYTES >> 3;                     // 8-byte elements
    u32x4 g0 = { 1u,                                  // count=1 valid descriptor
                 la,                                  // lds_addr
                 (unsigned)ga,                        // global_addr[31:0]
                 ((unsigned)(ga >> 32) & 0x1FFFFFFu) | (2u << 30) };  // addr[56:32] | type=2
    i32x8 g1 = { (int)(3u << 16),                     // data_size=8B, mask=0
                 (int)((E & 0xFFFFu) << 16),          // tensor_dim0[15:0]
                 (int)((E >> 16) | (1u << 16)),       // tensor_dim0[31:16] | tensor_dim1=1
                 (int)((E & 0xFFFFu) << 16),          // tile_dim0 = E
                 1,                                   // tile_dim1 = 1
                 (int)E,                              // tensor_dim0_stride
                 0, 0 };
    i32x4 z4 = { 0, 0, 0, 0 };
    i32x8 z8 = { 0, 0, 0, 0, 0, 0, 0, 0 };
    __builtin_amdgcn_tensor_load_to_lds(g0, g1, z4, z4, z8, 0);
#if __has_builtin(__builtin_amdgcn_s_wait_tensorcnt)
    __builtin_amdgcn_s_wait_tensorcnt(0);
#else
    asm volatile("s_wait_tensorcnt 0" ::: "memory");
#endif
  }
  __syncthreads();
#else
  const f4* src = (const f4*)gsrc;
  f4* dst = (f4*)wbuf;
  for (int i = threadIdx.x; i < WREG_BYTES / 16; i += blockDim.x) dst[i] = src[i];
  __syncthreads();
#endif
}

// one LSTM step for a 16-row batch tile; h of each gate tile is stored to the LDS
// stage as it is produced (no hreg array -> lower VGPR pressure); creg updated in place.
template <bool WRITE_GLOBAL>
__device__ __forceinline__ void lstm_step(v16h xA0, v16h xA1, v16h hA0, v16h hA1,
                                          const _Float16* wl_ih, const _Float16* wl_hh,
                                          const float* bl, float creg[4][8],
                                          _Float16* S, float* h_ws, float* c_ws,
                                          int m0, int lane) {
  int nlo = lane & 15, hi8 = (lane >> 4) << 3;
  #pragma unroll
  for (int jt = 0; jt < 4; ++jt) {
    v8f g[4];
    #pragma unroll
    for (int gt = 0; gt < 4; ++gt) {
      int nt = gt * 4 + jt;
      float bb = bl[nt * 16 + nlo];
      v8f acc = {bb, bb, bb, bb, bb, bb, bb, bb};
      acc = wmma32(xA0, load_bfrag(wl_ih + (nt * 2 + 0) * 512, lane), acc);
      acc = wmma32(xA1, load_bfrag(wl_ih + (nt * 2 + 1) * 512, lane), acc);
      acc = wmma32(hA0, load_bfrag(wl_hh + (nt * 2 + 0) * 512, lane), acc);
      acc = wmma32(hA1, load_bfrag(wl_hh + (nt * 2 + 1) * 512, lane), acc);
      g[gt] = acc;
    }
    int col = jt * 16 + nlo;
    #pragma unroll
    for (int e = 0; e < 8; ++e) {
      float ig = sigm(g[0][e]);
      float fg = sigm(g[1][e]);
      float gg = tanhf(g[2][e]);
      float og = sigm(g[3][e]);
      float cn = fg * creg[jt][e] + ig * gg;
      creg[jt][e] = cn;
      float hn = og * tanhf(cn);
      S[(hi8 + e) * 64 + col] = (_Float16)hn;
      if (WRITE_GLOBAL) {
        size_t o = (size_t)(m0 + hi8 + e) * 64 + col;
        h_ws[o] = hn;
        c_ws[o] = cn;
      }
    }
  }
}

// ---------------------------------------------------------------- kernels

// one-time: repack weights to WMMA B-fragment layout, fuse LSTM biases, zero stats
__global__ void pack_kernel(const float* Wih, const float* Whh, const float* seW2,
                            const float* hpW1, const float* bih, const float* bhh,
                            _Float16* wfih, _Float16* wfhh, _Float16* wfse2,
                            _Float16* wfhp1, float* blstm, float* stats) {
  int tid = threadIdx.x;
  for (int idx = tid; idx < 32 * 512; idx += blockDim.x) {
    int frag = idx >> 9, w = idx & 511;
    int lane = w >> 4, p = w & 15;
    int nt = frag >> 1, kc = frag & 1;
    int n = nt * 16 + (lane & 15);
    int k = kc * 32 + ((p >> 3) << 4) + ((lane >> 4) << 3) + (p & 7);
    wfih[idx] = (_Float16)Wih[n * 64 + k];
    wfhh[idx] = (_Float16)Whh[n * 64 + k];
  }
  for (int idx = tid; idx < 4 * 512; idx += blockDim.x) {
    int frag = idx >> 9, w = idx & 511;
    int lane = w >> 4, p = w & 15;
    int n = frag * 16 + (lane & 15);
    int k = ((p >> 3) << 4) + ((lane >> 4) << 3) + (p & 7);   // K padded 16->32 with zeros
    wfse2[idx] = (k < 16) ? (_Float16)seW2[n * 16 + k] : (_Float16)0.f;
  }
  for (int idx = tid; idx < 2 * 512; idx += blockDim.x) {
    int frag = idx >> 9, w = idx & 511;
    int lane = w >> 4, p = w & 15;
    int n = lane & 15;
    int k = frag * 32 + ((p >> 3) << 4) + ((lane >> 4) << 3) + (p & 7);
    wfhp1[idx] = (_Float16)hpW1[n * 64 + k];
  }
  for (int j = tid; j < 256; j += blockDim.x) blstm[j] = bih[j] + bhh[j];
  for (int j = tid; j < 128; j += blockDim.x) stats[j] = 0.f;
}

// encoder BN stats over T*B rows of spatial-embedding layer1; also init last_pos
__global__ void enc_stats_kernel(const float* obs, const float* seW1, const float* seb1,
                                 float* se_sum, float* se_sq, float* lp,
                                 const int* xm, const int* ym, int Bsz) {
  __shared__ float bs[16], bq[16];
  int tid = threadIdx.x;
  if (tid < 16) { bs[tid] = 0.f; bq[tid] = 0.f; }
  __syncthreads();
  int idx = blockIdx.x * blockDim.x + tid;
  if (idx < TT * Bsz) {
    float inv_x = 1.f / (float)(*xm), inv_y = 1.f / (float)(*ym);
    float ox = obs[(size_t)idx * 2] * inv_x, oy = obs[(size_t)idx * 2 + 1] * inv_y;
    #pragma unroll
    for (int kk = 0; kk < 16; ++kk) {
      int k = (kk + tid) & 15;   // rotate to spread LDS bank atomics
      float v = ox * seW1[2 * k] + oy * seW1[2 * k + 1] + seb1[k];
      atomicAdd(&bs[k], v);
      atomicAdd(&bq[k], v * v);
    }
    if (idx >= (TT - 1) * Bsz) {
      int b = idx - (TT - 1) * Bsz;
      lp[(size_t)b * 2 + 0] = ox;
      lp[(size_t)b * 2 + 1] = oy;
    }
  }
  __syncthreads();
  if (tid < 16) { atomicAdd(&se_sum[tid], bs[tid]); atomicAdd(&se_sq[tid], bq[tid]); }
}

// finalize BN: scale/shift from sums, then re-zero accumulators for next use
__global__ void fin_kernel(float* sum, float* sq, const float* gamma, const float* beta,
                           float* scale, float* shift, float invN) {
  int i = threadIdx.x;
  if (i < 16) {
    float mu  = sum[i] * invN;
    float var = sq[i] * invN - mu * mu;
    float sc  = gamma[i] * rsqrtf(var + BN_EPS);
    scale[i] = sc;
    shift[i] = beta[i] - mu * sc;
    sum[i] = 0.f;
    sq[i]  = 0.f;
  }
}

// encoder: fused embed + 8 LSTM steps, c in registers, h round-trips LDS stage only
__global__ __launch_bounds__(TPB, 1)
void enc_kernel(const float* obs, const float* seW1, const float* seb1,
                const float* seScale, const float* seShift, const float* seb2,
                const _Float16* wsbase, float* h_ws, float* c_ws,
                const int* xm, const int* ym, int Bsz) {
  __shared__ __align__(16) _Float16 wbuf[WREG_HALVES];
  __shared__ __align__(16) _Float16 stage[WAVES][16 * 64];
  load_weights_to_lds(wsbase, wbuf);
  const _Float16* wl_ih  = wbuf;
  const _Float16* wl_hh  = wbuf + 16384;
  const _Float16* wl_se2 = wbuf + 32768;
  const float*    bl     = (const float*)((const char*)wbuf + 71680);

  int lane = threadIdx.x & 31, wave = threadIdx.x >> 5;
  int m0 = (blockIdx.x * WAVES + wave) * 16;
  if (m0 >= Bsz) return;
  int nlo = lane & 15, hi8 = (lane >> 4) << 3;
  _Float16* S = stage[wave];
  float inv_x = 1.f / (float)(*xm), inv_y = 1.f / (float)(*ym);
  float creg[4][8];
  #pragma unroll
  for (int jt = 0; jt < 4; ++jt)
    #pragma unroll
    for (int e = 0; e < 8; ++e) creg[jt][e] = 0.f;
  v16h hA0 = {};   // h0 == 0
  v16h hA1 = {};
  for (int t = 0; t < TT; ++t) {
    const float* op = obs + ((size_t)t * Bsz + m0 + nlo) * 2;
    if (t + 1 < TT) __builtin_prefetch(op + (size_t)Bsz * 2, 0, 3);
    float ox = op[0] * inv_x, oy = op[1] * inv_y;
    v16h aS;   // layer1 + BN + relu built directly in A-fragment form (K padded to 32)
    #pragma unroll
    for (int i = 0; i < 8; ++i) {
      int k = hi8 + i;
      float v = ox * seW1[2 * k] + oy * seW1[2 * k + 1] + seb1[k];
      v = v * seScale[k] + seShift[k];
      aS[i]     = (_Float16)fmaxf(v, 0.f);
      aS[8 + i] = (_Float16)0.f;
    }
    wave_lds_fence();   // previous iteration's stage reads done before overwrite
    #pragma unroll
    for (int nt = 0; nt < 4; ++nt) {
      float bb = seb2[nt * 16 + nlo];
      v8f acc = {bb, bb, bb, bb, bb, bb, bb, bb};
      acc = wmma32(aS, load_bfrag(wl_se2 + nt * 512, lane), acc);
      store_tile_h16(S, nt, acc, lane);
    }
    wave_lds_fence();
    v16h xA0 = loadA_h16(S, 0, lane);
    v16h xA1 = loadA_h16(S, 32, lane);
    if (t == TT - 1)
      lstm_step<true >(xA0, xA1, hA0, hA1, wl_ih, wl_hh, bl, creg, S, h_ws, c_ws, m0, lane);
    else
      lstm_step<false>(xA0, xA1, hA0, hA1, wl_ih, wl_hh, bl, creg, S, h_ws, c_ws, m0, lane);
    wave_lds_fence();
    hA0 = loadA_h16(S, 0, lane);
    hA1 = loadA_h16(S, 32, lane);
  }
}

// decode pass A: h1 = h @ hpW1^T + b1 (WMMA), store + accumulate BN stats
__global__ __launch_bounds__(TPB, 1)
void hpA_kernel(const float* h_ws, const _Float16* wfhp1, const float* hpb1,
                float* h1b, float* hp_sum, float* hp_sq, int Bsz) {
  __shared__ float bs[16], bq[16];
  int tid = threadIdx.x;
  int lane = tid & 31, wave = tid >> 5;
  if (tid < 16) { bs[tid] = 0.f; bq[tid] = 0.f; }
  __syncthreads();
  int m0 = (blockIdx.x * WAVES + wave) * 16;
  if (m0 < Bsz) {
    int nlo = lane & 15, hi8 = (lane >> 4) << 3;
    v16h hA0 = loadA_f32(h_ws, 64, m0, 0, lane);
    v16h hA1 = loadA_f32(h_ws, 64, m0, 32, lane);
    float bb = hpb1[nlo];
    v8f acc = {bb, bb, bb, bb, bb, bb, bb, bb};
    acc = wmma32(hA0, load_bfrag(wfhp1, lane), acc);
    acc = wmma32(hA1, load_bfrag(wfhp1 + 512, lane), acc);
    float ls = 0.f, lq = 0.f;
    #pragma unroll
    for (int r = 0; r < 8; ++r) {
      float v = acc[r];
      h1b[(size_t)(m0 + hi8 + r) * 16 + nlo] = v;
      ls += v;
      lq += v * v;
    }
    atomicAdd(&bs[nlo], ls);
    atomicAdd(&bq[nlo], lq);
  }
  __syncthreads();
  if (tid < 16) { atomicAdd(&hp_sum[tid], bs[tid]); atomicAdd(&hp_sq[tid], bq[tid]); }
}

// decode pass B: bn+relu -> hidden2pos layer2 -> sigmoid pos update -> output,
// then se layer1 + BN stats for this step
__global__ void posB_kernel(const float* h1b, const float* hpScale, const float* hpShift,
                            const float* hpW2, const float* hpb2,
                            float* lp, const float* seW1, const float* seb1,
                            float* s1b, float* se_sum, float* se_sq,
                            float* outp, const int* xm, const int* ym, int Bsz) {
  __shared__ float bs[16], bq[16];
  int tid = threadIdx.x;
  if (tid < 16) { bs[tid] = 0.f; bq[tid] = 0.f; }
  __syncthreads();
  int r = blockIdx.x * blockDim.x + tid;
  if (r < Bsz) {
    float p0 = hpb2[0], p1 = hpb2[1];
    #pragma unroll
    for (int k = 0; k < 16; ++k) {
      float v = h1b[(size_t)r * 16 + k] * hpScale[k] + hpShift[k];
      v = fmaxf(v, 0.f);
      p0 += v * hpW2[k];
      p1 += v * hpW2[16 + k];
    }
    float l0 = sigm(p0 + lp[(size_t)r * 2 + 0]);
    float l1 = sigm(p1 + lp[(size_t)r * 2 + 1]);
    lp[(size_t)r * 2 + 0] = l0;
    lp[(size_t)r * 2 + 1] = l1;
    outp[(size_t)r * 2 + 0] = l0 * (float)(*xm);
    outp[(size_t)r * 2 + 1] = l1 * (float)(*ym);
    #pragma unroll
    for (int kk = 0; kk < 16; ++kk) {
      int k = (kk + tid) & 15;
      float v = l0 * seW1[2 * k] + l1 * seW1[2 * k + 1] + seb1[k];
      s1b[(size_t)r * 16 + k] = v;
      atomicAdd(&bs[k], v);
      atomicAdd(&bq[k], v * v);
    }
  }
  __syncthreads();
  if (tid < 16) { atomicAdd(&se_sum[tid], bs[tid]); atomicAdd(&se_sq[tid], bq[tid]); }
}

// decode pass C: x = relu(bn(s1)) @ seW2^T, then one LSTM step, h/c in place
__global__ __launch_bounds__(TPB, 1)
void stepC_kernel(const float* s1b, const float* seScale, const float* seShift,
                  const float* seb2, const _Float16* wsbase, const float* blstm_g,
                  float* h_ws, float* c_ws, int Bsz) {
  __shared__ __align__(16) _Float16 wbuf[WREG_HALVES];
  __shared__ __align__(16) _Float16 stage[WAVES][16 * 64];
  load_weights_to_lds(wsbase, wbuf);
  const _Float16* wl_ih  = wbuf;
  const _Float16* wl_hh  = wbuf + 16384;
  const _Float16* wl_se2 = wbuf + 32768;
  const float*    bl     = (const float*)((const char*)wbuf + 71680);

  int lane = threadIdx.x & 31, wave = threadIdx.x >> 5;
  int m0 = (blockIdx.x * WAVES + wave) * 16;
  if (m0 >= Bsz) return;
  int nlo = lane & 15, hi8 = (lane >> 4) << 3;
  _Float16* S = stage[wave];
  v16h aS;
  {
    const float* p = s1b + (size_t)(m0 + nlo) * 16 + hi8;
    #pragma unroll
    for (int i = 0; i < 8; ++i) {
      int k = hi8 + i;
      float v = p[i] * seScale[k] + seShift[k];
      aS[i]     = (_Float16)fmaxf(v, 0.f);
      aS[8 + i] = (_Float16)0.f;
    }
  }
  #pragma unroll
  for (int nt = 0; nt < 4; ++nt) {
    float bb = seb2[nt * 16 + nlo];
    v8f acc = {bb, bb, bb, bb, bb, bb, bb, bb};
    acc = wmma32(aS, load_bfrag(wl_se2 + nt * 512, lane), acc);
    store_tile_h16(S, nt, acc, lane);
  }
  wave_lds_fence();
  v16h xA0 = loadA_h16(S, 0, lane);
  v16h xA1 = loadA_h16(S, 32, lane);
  v16h hA0 = loadA_f32(h_ws, 64, m0, 0, lane);
  v16h hA1 = loadA_f32(h_ws, 64, m0, 32, lane);
  float creg[4][8];
  #pragma unroll
  for (int jt = 0; jt < 4; ++jt)
    #pragma unroll
    for (int r = 0; r < 8; ++r)
      creg[jt][r] = c_ws[(size_t)(m0 + hi8 + r) * 64 + jt * 16 + nlo];
  lstm_step<true>(xA0, xA1, hA0, hA1, wl_ih, wl_hh, bl, creg, S, h_ws, c_ws, m0, lane);
}

// ---------------------------------------------------------------- launch

extern "C" void kernel_launch(void* const* d_in, const int* in_sizes, int n_in,
                              void* d_out, int out_size, void* d_ws, size_t ws_size,
                              hipStream_t stream) {
  const float* obs  = (const float*)d_in[0];
  const float* seW1 = (const float*)d_in[1];
  const float* seb1 = (const float*)d_in[2];
  const float* seG  = (const float*)d_in[3];
  const float* seB  = (const float*)d_in[4];
  const float* seW2 = (const float*)d_in[5];
  const float* seb2 = (const float*)d_in[6];
  const float* hpW1 = (const float*)d_in[7];
  const float* hpb1 = (const float*)d_in[8];
  const float* hpG  = (const float*)d_in[9];
  const float* hpB  = (const float*)d_in[10];
  const float* hpW2 = (const float*)d_in[11];
  const float* hpb2 = (const float*)d_in[12];
  const float* Wih  = (const float*)d_in[13];
  const float* Whh  = (const float*)d_in[14];
  const float* bih  = (const float*)d_in[15];
  const float* bhh  = (const float*)d_in[16];
  const int* xm = (const int*)d_in[17];
  const int* ym = (const int*)d_in[18];
  int Bsz = in_sizes[0] / (TT * 2);
  float* out = (float*)d_out;

  char* ws = (char*)d_ws;
  _Float16* wfih  = (_Float16*)(ws);
  _Float16* wfhh  = (_Float16*)(ws + 32768);
  _Float16* wfse2 = (_Float16*)(ws + 65536);
  _Float16* wfhp1 = (_Float16*)(ws + 69632);
  float* blstm = (float*)(ws + 71680);
  float* stats = (float*)(ws + 72704);
  float* hp_sum = stats,        *hp_sq = stats + 16;
  float* hp_scale = stats + 32, *hp_shift = stats + 48;
  float* se_sum = stats + 64,   *se_sq = stats + 80;
  float* se_scale = stats + 96, *se_shift = stats + 112;
  float* h_ws = (float*)(ws + (1 << 20));
  float* c_ws = h_ws + (size_t)Bsz * 64;
  float* h1b  = c_ws + (size_t)Bsz * 64;
  float* s1b  = h1b + (size_t)Bsz * 16;
  float* lp   = s1b + (size_t)Bsz * 16;

  int tiles = (Bsz + 15) / 16;
  dim3 wblk(TPB);
  dim3 wgrid((tiles + WAVES - 1) / WAVES);

  pack_kernel<<<1, 256, 0, stream>>>(Wih, Whh, seW2, hpW1, bih, bhh,
                                     wfih, wfhh, wfse2, wfhp1, blstm, stats);
  enc_stats_kernel<<<(TT * Bsz + 255) / 256, 256, 0, stream>>>(
      obs, seW1, seb1, se_sum, se_sq, lp, xm, ym, Bsz);
  fin_kernel<<<1, 16, 0, stream>>>(se_sum, se_sq, seG, seB, se_scale, se_shift,
                                   1.f / (float)(TT * Bsz));
  enc_kernel<<<wgrid, wblk, 0, stream>>>(obs, seW1, seb1, se_scale, se_shift, seb2,
                                         wfih, h_ws, c_ws, xm, ym, Bsz);
  for (int s = 0; s < PRED; ++s) {
    hpA_kernel<<<wgrid, wblk, 0, stream>>>(h_ws, wfhp1, hpb1, h1b, hp_sum, hp_sq, Bsz);
    fin_kernel<<<1, 16, 0, stream>>>(hp_sum, hp_sq, hpG, hpB, hp_scale, hp_shift,
                                     1.f / (float)Bsz);
    posB_kernel<<<(Bsz + 255) / 256, 256, 0, stream>>>(
        h1b, hp_scale, hp_shift, hpW2, hpb2, lp, seW1, seb1, s1b,
        se_sum, se_sq, out + (size_t)s * Bsz * 2, xm, ym, Bsz);
    fin_kernel<<<1, 16, 0, stream>>>(se_sum, se_sq, seG, seB, se_scale, se_shift,
                                     1.f / (float)Bsz);
    stepC_kernel<<<wgrid, wblk, 0, stream>>>(s1b, se_scale, se_shift, seb2,
                                             wfih, blstm, h_ws, c_ws, Bsz);
  }
}
